// LowRankPants_43224550867813
// MI455X (gfx1250) — matrix operand order repaired and checked by reading, not compiled, not run on hardware
//
#include <hip/hip_runtime.h>
#include <stdint.h>

// ---------------------------------------------------------------------------
// B=8192, I=2048, O=256, N=64
// logits_flat[b, j] = sum_i x[b,i] * W[j%256, j/256, i] + bias[(j%256)*64 + j/256]
// encoded[b,o]      = softmax-weighted grid over logits[b, o*64 .. o*64+63]
// ---------------------------------------------------------------------------
#define B_DIM  8192
#define I_DIM  2048
#define O_DIM  256
#define N_BINS 64
#define J_DIM  (O_DIM * N_BINS)   // 16384

#define BM   128     // workgroup M tile
#define BJ   128     // workgroup J tile
#define KC   32      // K chunk == bf16 WMMA K
#define NKC  (I_DIM / KC)         // 64
#define LDST 80      // padded LDS row stride in bytes (32 bf16 = 64B + 16B pad)

typedef __attribute__((ext_vector_type(16))) __bf16 v16bf;
typedef __attribute__((ext_vector_type(8)))  float  v8f;
typedef __attribute__((ext_vector_type(4)))  int    v4i;

// address-space-qualified element types for the async-DMA builtin
typedef __attribute__((address_space(1))) v4i gv4i;   // global
typedef __attribute__((address_space(3))) v4i sv4i;   // LDS

struct Q2 { uint4 a, b; };   // 32 bytes == one v16bf fragment

// Split 8 f32 into packed bf16 hi / bf16 lo (Markidis split for ~fp32 GEMM).
__device__ __forceinline__ void cvt8(float4 f0, float4 f1, uint4 &h, uint4 &l) {
  float f[8] = {f0.x, f0.y, f0.z, f0.w, f1.x, f1.y, f1.z, f1.w};
  unsigned hb[8], lb[8];
#pragma unroll
  for (int i = 0; i < 8; ++i) {
    __bf16 hi = (__bf16)f[i];
    float  re = f[i] - (float)hi;
    __bf16 lo = (__bf16)re;
    hb[i] = (unsigned)__builtin_bit_cast(unsigned short, hi);
    lb[i] = (unsigned)__builtin_bit_cast(unsigned short, lo);
  }
  h = make_uint4(hb[0] | (hb[1] << 16), hb[2] | (hb[3] << 16),
                 hb[4] | (hb[5] << 16), hb[6] | (hb[7] << 16));
  l = make_uint4(lb[0] | (lb[1] << 16), lb[2] | (lb[3] << 16),
                 lb[4] | (lb[5] << 16), lb[6] | (lb[7] << 16));
}

__device__ __forceinline__ v16bf ldFragA(const unsigned char* base) {
  // A layout (16-bit 16x32): kh=0 -> K0..7 & K16..23 ; kh=1 -> K8..15 & K24..31
  Q2 q;
  q.a = *(const uint4*)(base);
  q.b = *(const uint4*)(base + 32);
  return __builtin_bit_cast(v16bf, q);
}
__device__ __forceinline__ v16bf ldFragB(const unsigned char* base) {
  // B layout (16-bit 32x16): kh=0 -> K0..7 & K16..23 ; kh=1 -> K8..15 & K24..31
  Q2 q;
  q.a = *(const uint4*)(base);
  q.b = *(const uint4*)(base + 32);
  return __builtin_bit_cast(v16bf, q);
}

// 16-byte async DMA: global -> LDS, tracked by ASYNCcnt (CDNA5 path).
#define ASYNC16(gp, lp)                                                        \
  __builtin_amdgcn_global_load_async_to_lds_b128(                              \
      (gv4i*)(gp), (sv4i*)(lp), 0, 0)

// ---------------------------------------------------------------------------
// Prep pass: split f32 -> bf16 hi/lo ONCE, into K-chunked layout
// dst element (kc, row, kk) at byte ((kc*rows + row)*32 + kk)*2, so a
// (128-row x 32-K) tile is one contiguous 8KB block (perfect async copies).
// For W, the GEMM-column -> W-row permutation is applied here (permuteW=1).
// ---------------------------------------------------------------------------
__global__ __launch_bounds__(256)
void lrp_split_chunked(const float* __restrict__ src, unsigned char* __restrict__ dhi,
                       unsigned char* __restrict__ dlo, int rows, int permuteW) {
  const size_t tid = (size_t)blockIdx.x * blockDim.x + threadIdx.x;  // rows*NKC threads
  const int r  = (int)(tid % (size_t)rows);
  const int kc = (int)(tid / (size_t)rows);
  const int srow = permuteW ? (((r & (O_DIM - 1)) << 6) + (r >> 8)) : r;
  const float4* s = (const float4*)(src + (size_t)srow * I_DIM + kc * KC);
  uint4 h[4], l[4];
#pragma unroll
  for (int i = 0; i < 4; ++i) cvt8(s[2 * i], s[2 * i + 1], h[i], l[i]);
  uint4* ph = (uint4*)(dhi + ((size_t)kc * rows + r) * 64);
  uint4* pl = (uint4*)(dlo + ((size_t)kc * rows + r) * 64);
#pragma unroll
  for (int i = 0; i < 4; ++i) { ph[i] = h[i]; pl[i] = l[i]; }
}

// ---------------------------------------------------------------------------
// Main GEMM, pre-split inputs: zero VALU in the K-loop; tiles arrive via
// double-buffered GLOBAL_LOAD_ASYNC_TO_LDS_B128 overlapped with WMMA.
// ---------------------------------------------------------------------------
__global__ __launch_bounds__(256)
void lrp_gemm_async(const unsigned char* __restrict__ xsH, const unsigned char* __restrict__ xsL,
                    const unsigned char* __restrict__ wpH, const unsigned char* __restrict__ wpL,
                    const float* __restrict__ bias, float* __restrict__ logits) {
  __shared__ __align__(16) unsigned char sm[2][4][BM * LDST];   // 80KB, dbl-buffered

  const int t     = threadIdx.x;
  const int mBase = blockIdx.y * BM;
  const int jBase = blockIdx.x * BJ;

  // each thread owns two 16B slots of each of the 4 tile arrays (512 slots/array)
  const int r0 = t >> 2,          s0 = (t & 3) * 16;
  const int r1 = (t + 256) >> 2,  s1 = ((t + 256) & 3) * 16;
  const size_t aOff0 = (size_t)(mBase + r0) * 64 + s0;
  const size_t aOff1 = (size_t)(mBase + r1) * 64 + s1;
  const size_t bOff0 = (size_t)(jBase + r0) * 64 + s0;
  const size_t bOff1 = (size_t)(jBase + r1) * 64 + s1;
  const size_t aStride = (size_t)B_DIM * 64;   // bytes per K-chunk plane
  const size_t bStride = (size_t)J_DIM * 64;
  const int l0 = r0 * LDST + s0, l1 = r1 * LDST + s1;

  auto issue = [&](int kc, int p) {
    const size_t ka = (size_t)kc * aStride, kb = (size_t)kc * bStride;
    ASYNC16(xsH + ka + aOff0, &sm[p][0][l0]);
    ASYNC16(xsH + ka + aOff1, &sm[p][0][l1]);
    ASYNC16(xsL + ka + aOff0, &sm[p][1][l0]);
    ASYNC16(xsL + ka + aOff1, &sm[p][1][l1]);
    ASYNC16(wpH + kb + bOff0, &sm[p][2][l0]);
    ASYNC16(wpH + kb + bOff1, &sm[p][2][l1]);
    ASYNC16(wpL + kb + bOff0, &sm[p][3][l0]);
    ASYNC16(wpL + kb + bOff1, &sm[p][3][l1]);
  };

  // compute mapping: 8 waves as 4(M) x 2(J); each wave owns 32x64
  const int lane = t & 31;
  const int wave = t >> 5;
  const int wm = (wave & 3) * 32;
  const int wj = (wave >> 2) * 64;
  const int lm = lane & 15;
  const int kh = lane >> 4;

  v8f acc[2][4] = {};

  issue(0, 0);
  asm volatile("s_wait_asynccnt 0" ::: "memory");
  __syncthreads();

  for (int kc = 0; kc < NKC; ++kc) {
    const int p = kc & 1;
    if (kc + 1 < NKC) issue(kc + 1, p ^ 1);     // DMA next chunk under compute

    v16bf aH[2], aL[2], bH[4], bL[4];
#pragma unroll
    for (int mi = 0; mi < 2; ++mi) {
      const int row = wm + mi * 16 + lm;
      aH[mi] = ldFragA(&sm[p][0][row * LDST + kh * 16]);
      aL[mi] = ldFragA(&sm[p][1][row * LDST + kh * 16]);
    }
#pragma unroll
    for (int ji = 0; ji < 4; ++ji) {
      const int col = wj + ji * 16 + lm;
      bH[ji] = ldFragB(&sm[p][2][col * LDST + kh * 16]);
      bL[ji] = ldFragB(&sm[p][3][col * LDST + kh * 16]);
    }
#pragma unroll
    for (int mi = 0; mi < 2; ++mi)
#pragma unroll
      for (int ji = 0; ji < 4; ++ji) {
        acc[mi][ji] = __builtin_amdgcn_wmma_f32_16x16x32_bf16(
            false, aH[mi], false, bH[ji], (short)0, acc[mi][ji], false, false);
        acc[mi][ji] = __builtin_amdgcn_wmma_f32_16x16x32_bf16(
            false, aH[mi], false, bL[ji], (short)0, acc[mi][ji], false, false);
        acc[mi][ji] = __builtin_amdgcn_wmma_f32_16x16x32_bf16(
            false, aL[mi], false, bH[ji], (short)0, acc[mi][ji], false, false);
      }

    asm volatile("s_wait_asynccnt 0" ::: "memory");  // next buffer landed
    __syncthreads();                                 // visible to all waves
  }

#pragma unroll
  for (int mi = 0; mi < 2; ++mi) {
#pragma unroll
    for (int ji = 0; ji < 4; ++ji) {
      const int col = jBase + wj + ji * 16 + lm;
      const int wr  = ((col & (O_DIM - 1)) << 6) + (col >> 8);
      const float bv = bias[wr];
      const int row0 = mBase + wm + mi * 16 + kh * 8;
      float* outp = logits + (size_t)row0 * J_DIM + col;
#pragma unroll
      for (int r = 0; r < 8; ++r)
        outp[(size_t)r * J_DIM] = acc[mi][ji][r] + bv;
    }
  }
}

// ---------------------------------------------------------------------------
// Fallback fused kernel (converts in-loop; used only if ws is too small).
// ---------------------------------------------------------------------------
__global__ __launch_bounds__(256)
void lrp_gemm_fused(const float* __restrict__ X, const float* __restrict__ W,
                    const float* __restrict__ bias, float* __restrict__ logits) {
  __shared__ __align__(16) unsigned char sAh[BM * LDST];
  __shared__ __align__(16) unsigned char sAl[BM * LDST];
  __shared__ __align__(16) unsigned char sBh[BJ * LDST];
  __shared__ __align__(16) unsigned char sBl[BJ * LDST];

  const int t     = threadIdx.x;
  const int mBase = blockIdx.y * BM;
  const int jBase = blockIdx.x * BJ;

  const int lr = t >> 1;
  const int lh = t & 1;
  const float* gA = X + (size_t)(mBase + lr) * I_DIM + lh * 16;
  const int jcol  = jBase + lr;
  const int wrowL = ((jcol & (O_DIM - 1)) << 6) + (jcol >> 8);
  const float* gB = W + (size_t)wrowL * I_DIM + lh * 16;
  unsigned char* stAh = sAh + lr * LDST + lh * 32;
  unsigned char* stAl = sAl + lr * LDST + lh * 32;
  unsigned char* stBh = sBh + lr * LDST + lh * 32;
  unsigned char* stBl = sBl + lr * LDST + lh * 32;

  const int lane = t & 31;
  const int wave = t >> 5;
  const int wm = (wave & 3) * 32;
  const int wj = (wave >> 2) * 64;
  const int lm = lane & 15;
  const int kh = lane >> 4;

  v8f acc[2][4] = {};

  for (int k0 = 0; k0 < I_DIM; k0 += KC) {
    if (k0 + KC < I_DIM) {
      __builtin_prefetch(gA + k0 + KC, 0, 1);
      __builtin_prefetch(gB + k0 + KC, 0, 1);
    }
    const float4* pa = (const float4*)(gA + k0);
    float4 a0 = pa[0], a1 = pa[1], a2 = pa[2], a3 = pa[3];
    const float4* pb = (const float4*)(gB + k0);
    float4 b0 = pb[0], b1 = pb[1], b2 = pb[2], b3 = pb[3];

    uint4 ah0, al0, ah1, al1, bh0, bl0, bh1, bl1;
    cvt8(a0, a1, ah0, al0);  cvt8(a2, a3, ah1, al1);
    cvt8(b0, b1, bh0, bl0);  cvt8(b2, b3, bh1, bl1);

    __syncthreads();
    ((uint4*)stAh)[0] = ah0;  ((uint4*)stAh)[1] = ah1;
    ((uint4*)stAl)[0] = al0;  ((uint4*)stAl)[1] = al1;
    ((uint4*)stBh)[0] = bh0;  ((uint4*)stBh)[1] = bh1;
    ((uint4*)stBl)[0] = bl0;  ((uint4*)stBl)[1] = bl1;
    __syncthreads();

    v16bf aH[2], aL[2], bH[4], bL[4];
#pragma unroll
    for (int mi = 0; mi < 2; ++mi) {
      const int row = wm + mi * 16 + lm;
      aH[mi] = ldFragA(sAh + row * LDST + kh * 16);
      aL[mi] = ldFragA(sAl + row * LDST + kh * 16);
    }
#pragma unroll
    for (int ji = 0; ji < 4; ++ji) {
      const int col = wj + ji * 16 + lm;
      bH[ji] = ldFragB(sBh + col * LDST + kh * 16);
      bL[ji] = ldFragB(sBl + col * LDST + kh * 16);
    }
#pragma unroll
    for (int mi = 0; mi < 2; ++mi)
#pragma unroll
      for (int ji = 0; ji < 4; ++ji) {
        acc[mi][ji] = __builtin_amdgcn_wmma_f32_16x16x32_bf16(
            false, aH[mi], false, bH[ji], (short)0, acc[mi][ji], false, false);
        acc[mi][ji] = __builtin_amdgcn_wmma_f32_16x16x32_bf16(
            false, aH[mi], false, bL[ji], (short)0, acc[mi][ji], false, false);
        acc[mi][ji] = __builtin_amdgcn_wmma_f32_16x16x32_bf16(
            false, aL[mi], false, bH[ji], (short)0, acc[mi][ji], false, false);
      }
  }

#pragma unroll
  for (int mi = 0; mi < 2; ++mi) {
#pragma unroll
    for (int ji = 0; ji < 4; ++ji) {
      const int col = jBase + wj + ji * 16 + lm;
      const int wr  = ((col & (O_DIM - 1)) << 6) + (col >> 8);
      const float bv = bias[wr];
      const int row0 = mBase + wm + mi * 16 + kh * 8;
      float* outp = logits + (size_t)row0 * J_DIM + col;
#pragma unroll
      for (int r = 0; r < 8; ++r)
        outp[(size_t)r * J_DIM] = acc[mi][ji][r] + bv;
    }
  }
}

// ---------------------------------------------------------------------------
// Gumbel-softmax + grid collapse: one wave32 per (b,o) group of 64 bins.
// ---------------------------------------------------------------------------
__global__ __launch_bounds__(256)
void lrp_gumbel_encode(const float* __restrict__ logits, const float* __restrict__ gnoise,
                       const float* __restrict__ grid, float* __restrict__ encoded,
                       int groups) {
  const int wid  = (int)((blockIdx.x * blockDim.x + threadIdx.x) >> 5);
  const int lane = threadIdx.x & 31;
  if (wid >= groups) return;
  const float* lp = logits + (size_t)wid * N_BINS;
  const float* gp = gnoise + (size_t)wid * N_BINS;

  float z0 = (lp[lane]      + gp[lane])      * 2.0f;   // /tau, tau=0.5
  float z1 = (lp[lane + 32] + gp[lane + 32]) * 2.0f;

  float m = fmaxf(z0, z1);
#pragma unroll
  for (int off = 16; off; off >>= 1) m = fmaxf(m, __shfl_xor(m, off, 32));

  float e0 = __expf(z0 - m), e1 = __expf(z1 - m);
  float s  = e0 + e1;
  float w  = e0 * grid[lane] + e1 * grid[lane + 32];
#pragma unroll
  for (int off = 16; off; off >>= 1) {
    s += __shfl_xor(s, off, 32);
    w += __shfl_xor(w, off, 32);
  }
  if (lane == 0) encoded[wid] = w / s;
}

// ---------------------------------------------------------------------------
extern "C" void kernel_launch(void* const* d_in, const int* in_sizes, int n_in,
                              void* d_out, int out_size, void* d_ws, size_t ws_size,
                              hipStream_t stream) {
  (void)in_sizes; (void)n_in; (void)out_size;
  const float* x    = (const float*)d_in[0];
  const float* W    = (const float*)d_in[1];
  const float* bias = (const float*)d_in[2];
  const float* grid = (const float*)d_in[3];
  const float* gn   = (const float*)d_in[4];

  float* out     = (float*)d_out;
  float* encoded = out;                                  // [B, O]
  float* logits  = out + (size_t)B_DIM * O_DIM;          // [B, O, N] flat

  const size_t szX = (size_t)B_DIM * I_DIM * 2;          // 32MB per half
  const size_t szW = (size_t)J_DIM * I_DIM * 2;          // 64MB per half
  const size_t need = 2 * szX + 2 * szW;                 // 192MB

  dim3 gdim(J_DIM / BJ, B_DIM / BM);                     // 128 x 64 workgroups
  if (ws_size >= need) {
    unsigned char* xsH = (unsigned char*)d_ws;
    unsigned char* xsL = xsH + szX;
    unsigned char* wpH = xsL + szX;
    unsigned char* wpL = wpH + szW;
    lrp_split_chunked<<<(B_DIM * NKC) / 256, dim3(256), 0, stream>>>(x, xsH, xsL, B_DIM, 0);
    lrp_split_chunked<<<(J_DIM * NKC) / 256, dim3(256), 0, stream>>>(W, wpH, wpL, J_DIM, 1);
    lrp_gemm_async<<<gdim, dim3(256), 0, stream>>>(xsH, xsL, wpH, wpL, bias, logits);
  } else {
    lrp_gemm_fused<<<gdim, dim3(256), 0, stream>>>(x, W, bias, logits);
  }

  const int groups = B_DIM * O_DIM;                      // 2,097,152 wave-groups
  const int blocks = (groups * 32) / 256;                // 262,144 blocks
  lrp_gumbel_encode<<<blocks, dim3(256), 0, stream>>>(logits, gn, grid, encoded, groups);
}
